// Dynamic_System_34789235097693
// MI455X (gfx1250) — matrix operand order repaired and checked by reading, not compiled, not run on hardware
//
#include <hip/hip_runtime.h>

typedef __attribute__((ext_vector_type(16))) _Float16 v16h;
typedef __attribute__((ext_vector_type(8)))  _Float16 v8h;
typedef __attribute__((ext_vector_type(8)))  float    v8f;

#define HID 256
#define TILE 16   // samples per workgroup

// ---------------------------------------------------------------------------
// Prep: build f16 B-operand layouts in workspace ([n][k], k contiguous, so a
// WMMA B fragment per lane (col n = nt*16+lane%16, k = kb+16*(lane/16)..+15)
// is two contiguous 16B loads):
//   Wf[n*256+k]  = W2[k][n]   (B = W2   for forward-direction GEMMs)
//   Wb[n*256+k]  = W2[n][k]   (B = W2^T for backward-direction GEMMs)
// ---------------------------------------------------------------------------
__global__ __launch_bounds__(256) void prep_kernel(const float* __restrict__ W2,
                                                   _Float16* __restrict__ Wf,
                                                   _Float16* __restrict__ Wb) {
  int i = blockIdx.x * 256 + threadIdx.x;   // i = k*256 + n (coalesced read)
  float w = W2[i];
  int k = i >> 8, n = i & 255;
  Wb[i] = (_Float16)w;              // identical layout, just f16
  Wf[n * 256 + k] = (_Float16)w;    // transpose
}

// W1h[n*256+k] = W1[n][k] for n<12, 0 for n=12..15  (B = W1^T, N padded to 16)
__global__ __launch_bounds__(256) void prep_w1_kernel(const float* __restrict__ W1,
                                                      _Float16* __restrict__ W1h) {
  int n = blockIdx.x;               // 0..15
  int k = threadIdx.x;              // 0..255
  W1h[n * 256 + k] = (n < 12) ? (_Float16)W1[n * 256 + k] : (_Float16)0.f;
}

// ---------------------------------------------------------------------------
// Main kernel: one workgroup = 16 samples, 256 threads (8 wave32).
// ---------------------------------------------------------------------------
__global__ __launch_bounds__(256) void lnn_kernel(
    const float* __restrict__ g_td, const float* __restrict__ g_sd,
    const float* __restrict__ g_th, const float* __restrict__ g_z,
    const float* __restrict__ g_s,  const float* __restrict__ g_sdd,
    const float* __restrict__ g_tau,
    const float* __restrict__ W1, const float* __restrict__ b1,
    const float* __restrict__ b2, const float* __restrict__ W3,
    const float* __restrict__ V1, const float* __restrict__ c1,
    const float* __restrict__ V2, const float* __restrict__ c2,
    const _Float16* __restrict__ Wf, const _Float16* __restrict__ Wb,
    const _Float16* __restrict__ W1h,
    float* __restrict__ out) {

  // LDS ~134KB -> 2 workgroups per WGP (320KB LDS on CDNA5)
  __shared__ _Float16 A16[64 * HID];        // 32KB: [h1|dh1_j] -> [u2|du2_j] -> [u1|du1_j]
  __shared__ float    scr1[TILE * HID];     // 16KB: a2, then v
  __shared__ float    scr3[3 * TILE * HID]; // 48KB: da2_j, then dv_j
  __shared__ float    t1s [TILE * HID];     // 16KB: 1 - h1^2
  __shared__ _Float16 m1s [TILE * HID];     //  8KB: -2*h1*t1
  __shared__ _Float16 hbs [TILE * HID];     //  8KB: B-net hidden tanh
  __shared__ float    gbuf[64 * 16];        //  4KB: [u1|du1_j] x W1^T result
  __shared__ float    xs  [TILE * 12];
  __shared__ float    xbs [TILE * 9];
  __shared__ float    taus[TILE * 3];
  __shared__ float    bos [TILE * 3];

  const int tid  = threadIdx.x;
  const int lane = tid & 31;
  const int wave = tid >> 5;
  const int mrow = lane & 15;   // A row / B column within a 16x16 tile
  const int grp  = lane >> 4;   // K half-group select (WMMA 16-bit layouts)
  const int base = blockIdx.x * TILE;

  // ---- phase 0: stage per-sample inputs -----------------------------------
  if (tid < TILE * 12) {
    int s = tid / 12, c = tid % 12;
    const float* p = (c < 3) ? g_td : (c < 6) ? g_sd : (c < 9) ? g_th : g_z;
    xs[tid] = p[(base + s) * 3 + (c % 3)];
  }
  if (tid < TILE * 9) {
    int s = tid / 9, c = tid % 9;
    const float* p = (c < 3) ? g_th : (c < 6) ? g_s : g_sdd;
    xbs[tid] = p[(base + s) * 3 + (c % 3)];
  }
  if (tid < TILE * 3) {
    int s = tid / 3, c = tid % 3;
    taus[tid] = g_tau[(base + s) * 3 + c];
  }
  __syncthreads();

  // ---- phase 1: layer-1 (K=12, VALU f32) + B-net hidden -------------------
  {
    float w1c[12];
#pragma unroll
    for (int c = 0; c < 12; ++c) w1c[c] = W1[c * HID + tid];
    float v1c[9];
#pragma unroll
    for (int c = 0; c < 9; ++c) v1c[c] = V1[c * HID + tid];
    float b1t = b1[tid], c1t = c1[tid];
#pragma unroll 4
    for (int s = 0; s < TILE; ++s) {
      float a = b1t;
#pragma unroll
      for (int c = 0; c < 12; ++c) a += xs[s * 12 + c] * w1c[c];
      float h  = tanhf(a);
      float t1 = 1.f - h * h;
      A16[s * HID + tid] = (_Float16)h;                 // h1 row
      t1s[s * HID + tid] = t1;
      m1s[s * HID + tid] = (_Float16)(-2.f * h * t1);
#pragma unroll
      for (int j = 0; j < 3; ++j)                       // tangent rows: t1 * W1[j,:]
        A16[(16 + 16 * j + s) * HID + tid] = (_Float16)(t1 * w1c[j]);
      float ab = c1t;
#pragma unroll
      for (int c = 0; c < 9; ++c) ab += xbs[s * 9 + c] * v1c[c];
      hbs[s * HID + tid] = (_Float16)tanhf(ab);
    }
  }
  __syncthreads();

  // ---- shared GEMM body: [64 x 256](LDS f16) x [256 x 256](global f16) ----
  // Per wave: N-tiles {wave, wave+8}. For each N-tile, the 8 K-step B
  // fragments are loaded ONCE into registers and reused across all 4 M-tiles
  // (4x fewer global B loads). Mtile0 -> scr1, Mtile(1+j) -> scr3[j].
  auto run_gemm = [&](const _Float16* __restrict__ Bp) {
    for (int nn = 0; nn < 2; ++nn) {
      const int nt = wave + 8 * nn;
      const _Float16* bp = Bp + (nt * 16 + mrow) * HID + 16 * grp;
      v16h bfrag[8];
#pragma unroll
      for (int kk = 0; kk < 8; ++kk) {
        v8h b0 = *(const v8h*)(bp + kk * 32);
        v8h b1v = *(const v8h*)(bp + kk * 32 + 8);
        bfrag[kk] = __builtin_shufflevector(b0, b1v,
            0,1,2,3,4,5,6,7,8,9,10,11,12,13,14,15);
      }
      for (int mt = 0; mt < 4; ++mt) {
        const _Float16* ap = &A16[(mt * 16 + mrow) * HID];
        v8f acc = {};
#pragma unroll
        for (int kk = 0; kk < 8; ++kk) {
          v8h a0 = *(const v8h*)(ap + kk * 32 + 16 * grp);
          v8h a1 = *(const v8h*)(ap + kk * 32 + 16 * grp + 8);
          v16h av = __builtin_shufflevector(a0, a1,
              0,1,2,3,4,5,6,7,8,9,10,11,12,13,14,15);
          acc = __builtin_amdgcn_wmma_f32_16x16x32_f16(false, av, false, bfrag[kk],
                                                       (short)0, acc, false, false);
        }
        float* dst = (mt == 0) ? &scr1[0] : &scr3[(mt - 1) * TILE * HID];
#pragma unroll
        for (int r = 0; r < 8; ++r)   // C elem r -> M = r + 8*grp, N = mrow
          dst[(grp * 8 + r) * HID + nt * 16 + mrow] = acc[r];
      }
    }
  };

  // ---- phase 2: forward GEMMs: a2 = h1*W2, da2_j = dh1_j*W2 ---------------
  run_gemm(Wf);
  __syncthreads();

  // ---- phase 3: layer-2 nonlinearity + tangent chain ----------------------
  {
    float w3k = W3[tid];
    float b2t = b2[tid];
#pragma unroll 4
    for (int s = 0; s < TILE; ++s) {
      float a2 = scr1[s * HID + tid] + b2t;
      float h2 = tanhf(a2);
      float t2 = 1.f - h2 * h2;
      A16[s * HID + tid] = (_Float16)(t2 * w3k);        // u2
      float q2 = -2.f * h2 * t2 * w3k;
#pragma unroll
      for (int j = 0; j < 3; ++j)                       // du2_j = q2 * da2_j
        A16[(16 + 16 * j + s) * HID + tid] =
            (_Float16)(q2 * scr3[(j * TILE + s) * HID + tid]);
    }
  }
  __syncthreads();

  // ---- phase 4: backward GEMMs: v = u2*W2^T, dv_j = du2_j*W2^T ------------
  run_gemm(Wb);
  __syncthreads();

  // ---- phase 5: u1 = t1*v ; du1_j = t1*dv_j + (-2 h1 t1 v)*W1[j,:] --------
  // write as f16 A rows for the final W1^T GEMM
  {
#pragma unroll 4
    for (int s = 0; s < TILE; ++s) {
      float vv = scr1[s * HID + tid];
      float t1 = t1s[s * HID + tid];
      A16[s * HID + tid] = (_Float16)(t1 * vv);                 // u1
      float p1 = (float)m1s[s * HID + tid] * vv;
#pragma unroll
      for (int j = 0; j < 3; ++j) {
        float rj  = W1[j * HID + tid];
        float dvv = scr3[(j * TILE + s) * HID + tid];
        A16[(16 + 16 * j + s) * HID + tid] = (_Float16)(t1 * dvv + p1 * rj);
      }
    }
  }
  __syncthreads();

  // ---- phase 6: [u1|du1_j](64x256) x W1^T(256x16) on WMMA (waves 0-3),
  //               B-net output dots on VALU (waves 4-7), in parallel ---------
  if (wave < 4) {
    const int mt = wave;                                 // one M-tile per wave
    const _Float16* ap = &A16[(mt * 16 + mrow) * HID];
    const _Float16* bp = W1h + mrow * HID + 16 * grp;    // nt = 0
    v8f acc = {};
#pragma unroll
    for (int kk = 0; kk < 8; ++kk) {
      v8h a0 = *(const v8h*)(ap + kk * 32 + 16 * grp);
      v8h a1 = *(const v8h*)(ap + kk * 32 + 16 * grp + 8);
      v8h b0 = *(const v8h*)(bp + kk * 32);
      v8h b1v = *(const v8h*)(bp + kk * 32 + 8);
      v16h av = __builtin_shufflevector(a0, a1,
          0,1,2,3,4,5,6,7,8,9,10,11,12,13,14,15);
      v16h bv = __builtin_shufflevector(b0, b1v,
          0,1,2,3,4,5,6,7,8,9,10,11,12,13,14,15);
      acc = __builtin_amdgcn_wmma_f32_16x16x32_f16(false, av, false, bv,
                                                   (short)0, acc, false, false);
    }
#pragma unroll
    for (int r = 0; r < 8; ++r)
      gbuf[(mt * 16 + grp * 8 + r) * 16 + mrow] = acc[r];
  } else {
    int task = tid - 128;                 // 48 tasks: b_out[s][c]
    if (task < 48) {
      int s = task / 3, c = task % 3;
      float acc = c2[c];
      const _Float16* hv = hbs + s * HID;
#pragma unroll 4
      for (int k = 0; k < HID; ++k) acc += (float)hv[k] * V2[k * 3 + c];
      bos[task] = acc;
    }
  }
  __syncthreads();

  // ---- phase 7: rhs = tau + b_out + g_th - C*td ; out = H^-1 rhs ----------
  // gbuf row s       = g_x[s]  (cols 6..8 = g_th)
  // gbuf row 16+16j+s= dg_j    (cols 0..2 -> H[:,j]-I, cols 6..8 -> C[:,j])
  if (tid < TILE) {
    int s = tid;
    float td0 = xs[s * 12 + 0], td1 = xs[s * 12 + 1], td2 = xs[s * 12 + 2];
    float H[3][3], C[3][3];
#pragma unroll
    for (int j = 0; j < 3; ++j) {
      const float* dg = &gbuf[(16 + 16 * j + s) * 16];
#pragma unroll
      for (int c = 0; c < 3; ++c) {
        H[c][j] = dg[c] + ((c == j) ? 1.f : 0.f);
        C[c][j] = dg[6 + c];
      }
    }
    const float* gx = &gbuf[s * 16];
    float r0 = taus[s*3+0] + bos[s*3+0] + gx[6] - (C[0][0]*td0 + C[0][1]*td1 + C[0][2]*td2);
    float r1 = taus[s*3+1] + bos[s*3+1] + gx[7] - (C[1][0]*td0 + C[1][1]*td1 + C[1][2]*td2);
    float r2 = taus[s*3+2] + bos[s*3+2] + gx[8] - (C[2][0]*td0 + C[2][1]*td1 + C[2][2]*td2);
    float co00 =  (H[1][1]*H[2][2] - H[1][2]*H[2][1]);
    float co01 = -(H[1][0]*H[2][2] - H[1][2]*H[2][0]);
    float co02 =  (H[1][0]*H[2][1] - H[1][1]*H[2][0]);
    float det  = H[0][0]*co00 + H[0][1]*co01 + H[0][2]*co02;
    float inv  = 1.f / det;
    float a00 =  co00*inv;
    float a01 = -(H[0][1]*H[2][2] - H[0][2]*H[2][1])*inv;
    float a02 =  (H[0][1]*H[1][2] - H[0][2]*H[1][1])*inv;
    float a10 =  co01*inv;
    float a11 =  (H[0][0]*H[2][2] - H[0][2]*H[2][0])*inv;
    float a12 = -(H[0][0]*H[1][2] - H[0][2]*H[1][0])*inv;
    float a20 =  co02*inv;
    float a21 = -(H[0][0]*H[2][1] - H[0][1]*H[2][0])*inv;
    float a22 =  (H[0][0]*H[1][1] - H[0][1]*H[1][0])*inv;
    float* op = out + (size_t)(base + s) * 3;
    op[0] = a00*r0 + a01*r1 + a02*r2;
    op[1] = a10*r0 + a11*r1 + a12*r2;
    op[2] = a20*r0 + a21*r1 + a22*r2;
  }
}

// ---------------------------------------------------------------------------
extern "C" void kernel_launch(void* const* d_in, const int* in_sizes, int n_in,
                              void* d_out, int out_size, void* d_ws, size_t ws_size,
                              hipStream_t stream) {
  const float* td  = (const float*)d_in[0];
  const float* sd  = (const float*)d_in[1];
  const float* th  = (const float*)d_in[2];
  const float* z   = (const float*)d_in[3];
  const float* s   = (const float*)d_in[4];
  const float* sdd = (const float*)d_in[5];
  const float* tau = (const float*)d_in[6];
  const float* W1  = (const float*)d_in[7];
  const float* b1  = (const float*)d_in[8];
  const float* W2  = (const float*)d_in[9];
  const float* b2  = (const float*)d_in[10];
  const float* W3  = (const float*)d_in[11];
  /* b3 = d_in[12] unused: only gradients of L are needed */
  const float* V1  = (const float*)d_in[13];
  const float* c1  = (const float*)d_in[14];
  const float* V2  = (const float*)d_in[15];
  const float* c2  = (const float*)d_in[16];

  _Float16* Wf  = (_Float16*)d_ws;          // [n][k] = W2[k][n]   (128KB)
  _Float16* Wb  = Wf + HID * HID;           // [n][k] = W2[n][k]   (128KB)
  _Float16* W1h = Wb + HID * HID;           // [n][k] = W1 padded  (8KB)

  prep_kernel<<<HID, 256, 0, stream>>>(W2, Wf, Wb);
  prep_w1_kernel<<<16, 256, 0, stream>>>(W1, W1h);

  int Bn = in_sizes[0] / 3;                 // 131072
  lnn_kernel<<<Bn / TILE, 256, 0, stream>>>(
      td, sd, th, z, s, sdd, tau,
      W1, b1, b2, W3, V1, c1, V2, c2,
      Wf, Wb, W1h, (float*)d_out);
}
